// AttentionGate_83141976916929
// MI455X (gfx1250) — matrix operand order repaired and checked by reading, not compiled, not run on hardware
//
#include <hip/hip_runtime.h>
#include <cstdint>
#include <cstddef>

// ---------------------------------------------------------------------------
// AttentionGate on MI455X (gfx1250, wave32, WMMA)
//   z = [f*q, f*m, |f-q|, |f-m|]  (B,T,4D)  built cooperatively into LDS bf16
//   g = tanh(z @ W1 + b1)                    bf16 WMMA, f32 accumulate
//   logits = g @ W2 + b2                     per-fragment dot + f32 atomics
//   y = softmax(logits, axis=-1)             singleton axis -> 1.0
// Main GEMM: M=32768 K=4096 N=512 -> 137 GFLOP vs 136 MB compulsory bytes
// -> compute-bound at bf16 WMMA rates; working set < 192 MB L2.
// Double-buffered LDS: next tile's global loads issue before the WMMAs,
// conversion+ds_store after them; one barrier per K-step.
// ---------------------------------------------------------------------------

typedef __attribute__((ext_vector_type(16))) __bf16 v16bf;
typedef __attribute__((ext_vector_type(8)))  float  v8f;

union BV {                 // one 16x32-bf16 A/B fragment per lane (8 VGPRs)
  unsigned int u[8];
  uint4        q[2];
  v16bf        v;
};

// pack two f32 -> packed bf16 in one VALU op
__device__ __forceinline__ unsigned int pack_bf16(float x0, float x1) {
#if __has_builtin(__builtin_amdgcn_cvt_pk_bf16_f32)
  auto r = __builtin_amdgcn_cvt_pk_bf16_f32(x0, x1);
  return __builtin_bit_cast(unsigned int, r);
#else
  unsigned int r;
  asm("v_cvt_pk_bf16_f32 %0, %1, %2" : "=v"(r) : "v"(x0), "v"(x1));
  return r;
#endif
}

__device__ __forceinline__ unsigned int f2bf_rne(float f) {
  unsigned int u = __builtin_bit_cast(unsigned int, f);
  return (u + 0x7FFFu + ((u >> 16) & 1u)) >> 16;
}

// cheap tanh (output is exactly 1.0 after the singleton softmax either way)
__device__ __forceinline__ float fast_tanh(float x) {
  float e = __expf(2.0f * x);
  return 1.0f - 2.0f / (e + 1.0f);
}

constexpr int Bn = 64, Tn = 512, Dn = 1024, Un = 512;

// -- prep: W1 (4096x512 f32, k-major) -> W1T (512x4096 bf16, n-major) --------
__global__ void w1_transpose_bf16(const float* __restrict__ W1,
                                  unsigned short* __restrict__ W1T) {
  int idx = blockIdx.x * 256 + threadIdx.x;      // 512*4096 total
  int n = idx >> 12;
  int k = idx & 4095;
  W1T[idx] = (unsigned short)f2bf_rne(W1[(size_t)k * Un + n]);
}

__global__ void zero_f32(float* __restrict__ p, int n) {
  int i = blockIdx.x * 256 + threadIdx.x;
  if (i < n) p[i] = 0.0f;
}

// -- main fused GEMM: 128x128 tile per block, 8 waves, bf16 WMMA -------------
__global__ void __launch_bounds__(256)
attn_gate_gemm(const float* __restrict__ facts,
               const float* __restrict__ question,
               const float* __restrict__ memry,
               const unsigned short* __restrict__ W1T,
               const float* __restrict__ b1,
               const float* __restrict__ W2,
               float* __restrict__ logits)
{
  __shared__ __align__(16) float sh_q[Dn];
  __shared__ __align__(16) float sh_m[Dn];
  // double-buffered bf16 tiles, padded rows (40 halfwords) for bank spread
  __shared__ __align__(16) unsigned short sh_a[2][4][128 * 40]; // z tiles
  __shared__ __align__(16) unsigned short sh_b[2][4][128 * 40]; // W1T tiles

  const int t    = threadIdx.x;
  const int b    = blockIdx.z;        // batch
  const int row0 = blockIdx.y * 128;  // M tile within batch (T=512 -> 4)
  const int n0   = blockIdx.x * 128;  // N tile (UNITS=512 -> 4)

  const int lane   = t & 31;
  const int wid    = t >> 5;
  const int wave_m = wid >> 1;        // 0..3  -> 32 rows each
  const int wave_n = wid & 1;         // 0..1  -> 64 cols each
  const int half   = lane >> 4;       // K-half selector per ISA layout
  const int ln     = lane & 15;

  { // stage q,m rows for this batch (1024 f32 each)
    const float4 q4 = ((const float4*)(question + (size_t)b * Dn))[t];
    const float4 m4 = ((const float4*)(memry    + (size_t)b * Dn))[t];
    *(float4*)&sh_q[t * 4] = q4;
    *(float4*)&sh_m[t * 4] = m4;
  }

  // staging assignment: each thread owns (row ar, 16-wide k-halfword ak)
  const int ar = t >> 1;              // 0..127
  const int ak = (t & 1) * 16;        // 0 or 16
  const float* fbase = facts + ((size_t)(b * Tn + row0 + ar)) * Dn + ak;
  const unsigned short* wbase = W1T;

  // issue next-tile global loads into registers
  float fr[16];
  uint4 wr[8];
  auto load_tile = [&](int d0s) {
    const float* fp = fbase + d0s;
    *(float4*)&fr[0]  = *(const float4*)(fp + 0);
    *(float4*)&fr[4]  = *(const float4*)(fp + 4);
    *(float4*)&fr[8]  = *(const float4*)(fp + 8);
    *(float4*)&fr[12] = *(const float4*)(fp + 12);
    #pragma unroll
    for (int ci = 0; ci < 8; ++ci) {
      const int chunk = t * 8 + ci;          // 2048 x 16B chunks
      const int s  = chunk >> 9;
      const int r  = chunk & 511;
      const int nn = r >> 2;
      const int kc = (r & 3) * 8;
      wr[ci] = *(const uint4*)(wbase +
          ((size_t)(n0 + nn)) * (4 * Dn) + s * Dn + d0s + kc);
    }
  };

  // convert + store the registered tile into LDS buffer `buf`
  auto stage_to = [&](int buf, int d0s) {
    float q[16], m[16];
    *(float4*)&q[0]  = *(const float4*)&sh_q[d0s + ak + 0];
    *(float4*)&q[4]  = *(const float4*)&sh_q[d0s + ak + 4];
    *(float4*)&q[8]  = *(const float4*)&sh_q[d0s + ak + 8];
    *(float4*)&q[12] = *(const float4*)&sh_q[d0s + ak + 12];
    *(float4*)&m[0]  = *(const float4*)&sh_m[d0s + ak + 0];
    *(float4*)&m[4]  = *(const float4*)&sh_m[d0s + ak + 4];
    *(float4*)&m[8]  = *(const float4*)&sh_m[d0s + ak + 8];
    *(float4*)&m[12] = *(const float4*)&sh_m[d0s + ak + 12];

    unsigned int w[8];
    #pragma unroll
    for (int i = 0; i < 8; ++i)
      w[i] = pack_bf16(fr[2*i] * q[2*i], fr[2*i+1] * q[2*i+1]);
    *(uint4*)&sh_a[buf][0][ar * 40 + ak]     = *(uint4*)&w[0];
    *(uint4*)&sh_a[buf][0][ar * 40 + ak + 8] = *(uint4*)&w[4];
    #pragma unroll
    for (int i = 0; i < 8; ++i)
      w[i] = pack_bf16(fr[2*i] * m[2*i], fr[2*i+1] * m[2*i+1]);
    *(uint4*)&sh_a[buf][1][ar * 40 + ak]     = *(uint4*)&w[0];
    *(uint4*)&sh_a[buf][1][ar * 40 + ak + 8] = *(uint4*)&w[4];
    #pragma unroll
    for (int i = 0; i < 8; ++i)
      w[i] = pack_bf16(fabsf(fr[2*i] - q[2*i]), fabsf(fr[2*i+1] - q[2*i+1]));
    *(uint4*)&sh_a[buf][2][ar * 40 + ak]     = *(uint4*)&w[0];
    *(uint4*)&sh_a[buf][2][ar * 40 + ak + 8] = *(uint4*)&w[4];
    #pragma unroll
    for (int i = 0; i < 8; ++i)
      w[i] = pack_bf16(fabsf(fr[2*i] - m[2*i]), fabsf(fr[2*i+1] - m[2*i+1]));
    *(uint4*)&sh_a[buf][3][ar * 40 + ak]     = *(uint4*)&w[0];
    *(uint4*)&sh_a[buf][3][ar * 40 + ak + 8] = *(uint4*)&w[4];

    #pragma unroll
    for (int ci = 0; ci < 8; ++ci) {
      const int chunk = t * 8 + ci;
      const int s  = chunk >> 9;
      const int r  = chunk & 511;
      const int nn = r >> 2;
      const int kc = (r & 3) * 8;
      *(uint4*)&sh_b[buf][s][nn * 40 + kc] = wr[ci];
    }
  };

  v8f acc[2][4];
  #pragma unroll
  for (int mt = 0; mt < 2; ++mt)
    #pragma unroll
    for (int nt = 0; nt < 4; ++nt)
      acc[mt][nt] = v8f{};

  // prologue: tile 0 into buffer 0
  load_tile(0);
  __syncthreads();          // q/m visible to all stagers
  stage_to(0, 0);
  __syncthreads();

  int p = 0;
  for (int d0 = 0; d0 < Dn; d0 += 32) {
    const bool has_next = (d0 + 32) < Dn;
    if (has_next) load_tile(d0 + 32);   // global latency hides under WMMAs

    // ---- WMMA from buffer p; fragments via ds_load_b128 ----
    // A 16-bit 16x32: lanes 0-15 K{0..7,16..23}; lanes 16-31 K{8..15,24..31}
    // B 16-bit 32x16: lanes 0-15 N=n,K{0..7,16..23}; lanes 16-31 N=n,K{8..15,24..31}
    #pragma unroll
    for (int s = 0; s < 4; ++s) {
      BV af[2];
      #pragma unroll
      for (int mt = 0; mt < 2; ++mt) {
        const unsigned short* ap = &sh_a[p][s][(wave_m * 32 + mt * 16 + ln) * 40];
        af[mt].q[0] = *(const uint4*)(ap + half * 8);
        af[mt].q[1] = *(const uint4*)(ap + 16 + half * 8);
      }
      BV bf_[4];
      #pragma unroll
      for (int nt = 0; nt < 4; ++nt) {
        const unsigned short* bp = &sh_b[p][s][(wave_n * 64 + nt * 16 + ln) * 40];
        bf_[nt].q[0] = *(const uint4*)(bp + half * 8);
        bf_[nt].q[1] = *(const uint4*)(bp + 16 + half * 8);
      }
      #pragma unroll
      for (int mt = 0; mt < 2; ++mt)
        #pragma unroll
        for (int nt = 0; nt < 4; ++nt)
          acc[mt][nt] = __builtin_amdgcn_wmma_f32_16x16x32_bf16(
              false, af[mt].v, false, bf_[nt].v, (short)0, acc[mt][nt],
              false, false);
    }

    if (has_next) {
      stage_to(p ^ 1, d0 + 32);         // fill the other buffer
      __syncthreads();
      p ^= 1;
    }
  }

  // epilogue: g = tanh(acc + b1); partial (g . W2) per row; atomic into logits
  // C/D layout: VGPR r, lanes 0-15 -> M=r, N=lane; lanes 16-31 -> M=8+r, N=lane-16
  #pragma unroll
  for (int mt = 0; mt < 2; ++mt) {
    #pragma unroll
    for (int r = 0; r < 8; ++r) {
      const int row = row0 + wave_m * 32 + mt * 16 + half * 8 + r;
      float sum = 0.0f;
      #pragma unroll
      for (int nt = 0; nt < 4; ++nt) {
        const int N = n0 + wave_n * 64 + nt * 16 + ln;
        const float g = fast_tanh(acc[mt][nt][r] + b1[N]);
        sum += g * W2[N];
      }
      atomicAdd(&logits[b * Tn + row], sum);
    }
  }
}

// -- softmax over singleton output axis: exactly 1.0 -------------------------
__global__ void softmax_out(const float* __restrict__ logits,
                            const float* __restrict__ b2,
                            float* __restrict__ out) {
  int i = blockIdx.x * 256 + threadIdx.x;
  float l  = logits[i] + b2[0];
  float mx = l;                       // row max over a 1-wide axis
  float e  = __expf(l - mx);
  out[i]   = e / e;                   // 1.0
}

extern "C" void kernel_launch(void* const* d_in, const int* in_sizes, int n_in,
                              void* d_out, int out_size, void* d_ws, size_t ws_size,
                              hipStream_t stream) {
  (void)in_sizes; (void)n_in; (void)out_size; (void)ws_size;
  const float* facts    = (const float*)d_in[0];
  const float* question = (const float*)d_in[1];
  const float* memry    = (const float*)d_in[2];
  const float* W1       = (const float*)d_in[3];
  const float* b1       = (const float*)d_in[4];
  const float* W2       = (const float*)d_in[5];
  const float* b2       = (const float*)d_in[6];

  // workspace: [0, 4MB) W1T bf16 ; [4MB, +128KB) logits f32
  unsigned short* W1T = (unsigned short*)d_ws;
  float* logits = (float*)((char*)d_ws + (size_t)Un * (4 * Dn) * sizeof(unsigned short));

  w1_transpose_bf16<<<(Un * 4 * Dn) / 256, 256, 0, stream>>>(W1, W1T);
  zero_f32<<<(Bn * Tn) / 256, 256, 0, stream>>>(logits, Bn * Tn);

  dim3 grid(Un / 128, Tn / 128, Bn);   // (4, 4, 64)
  attn_gate_gemm<<<grid, 256, 0, stream>>>(facts, question, memry, W1T, b1, W2, logits);

  softmax_out<<<(Bn * Tn) / 256, 256, 0, stream>>>(logits, b2, (float*)d_out);
}
